// MaSSA_11209864642922
// MI455X (gfx1250) — matrix operand order repaired and checked by reading, not compiled, not run on hardware
//
#include <hip/hip_runtime.h>
#include <hip/hip_bf16.h>

typedef __bf16 bf16;
typedef __attribute__((ext_vector_type(16))) __bf16 v16bf;
typedef __attribute__((ext_vector_type(8)))  __bf16 bf16x8;
typedef __attribute__((ext_vector_type(8)))  float  v8f;
typedef __attribute__((ext_vector_type(4)))  int    i32x4;

#define EMBED 768
#define RDIM  384
#define KLM   4
#define BATCH 8
#define SEQ   4096
#define MTOK  (BATCH*SEQ)   // 32768 tokens

// ---------------------------------------------------------------------------
// CDNA5 async memory->LDS path (ASYNCcnt), guarded so we always compile
// ---------------------------------------------------------------------------
#if __has_builtin(__builtin_amdgcn_global_load_async_to_lds_b128)
#define USE_ASYNC 1
#else
#define USE_ASYNC 0
#endif

#if USE_ASYNC
__device__ __forceinline__ void async_cp16(const bf16* g, bf16* l) {
  // builtin expects (int4 AS1*, int4 AS3*, imm offset, imm cpol)
  __builtin_amdgcn_global_load_async_to_lds_b128(
      (__attribute__((address_space(1))) i32x4*)(i32x4*)(void*)(bf16*)g,
      (__attribute__((address_space(3))) i32x4*)(i32x4*)(void*)l, 0, 0);
}
__device__ __forceinline__ void wait_async_le4() {
#if __has_builtin(__builtin_amdgcn_s_wait_asynccnt)
  __builtin_amdgcn_s_wait_asynccnt(4);
#else
  asm volatile("s_wait_asynccnt 0x4" ::: "memory");
#endif
}
__device__ __forceinline__ void wait_async_0() {
#if __has_builtin(__builtin_amdgcn_s_wait_asynccnt)
  __builtin_amdgcn_s_wait_asynccnt(0);
#else
  asm volatile("s_wait_asynccnt 0x0" ::: "memory");
#endif
}
#endif

// ---------------------------------------------------------------------------
// Elementwise fp32 -> bf16 conversion
// ---------------------------------------------------------------------------
__global__ void k_f32_to_bf16(const float* __restrict__ in, bf16* __restrict__ out, int n) {
  int i = blockIdx.x * 256 + threadIdx.x;
  if (i < n) out[i] = (bf16)in[i];
}

// Transpose + convert weight W[din,dout] (row-major) -> Wt[dout,din] bf16
__global__ void k_wt_cvt(const float* __restrict__ W, bf16* __restrict__ Wt, int din, int dout) {
  int i = blockIdx.x * 256 + threadIdx.x;
  if (i < din * dout) {
    int o = i / din, ii = i - o * din;
    Wt[i] = (bf16)W[(size_t)ii * dout + o];
  }
}

// ---------------------------------------------------------------------------
// I = x @ Wi + bi   ([MTOK, 4]) — tiny N, scalar VALU is the right tool
// ---------------------------------------------------------------------------
__global__ void k_proj_I(const float* __restrict__ x, const float* __restrict__ Wi,
                         const float* __restrict__ bi, float* __restrict__ I) {
  int idx = blockIdx.x * 256 + threadIdx.x;      // MTOK*4 threads
  if (idx >= MTOK * KLM) return;
  int n = idx >> 2, k = idx & 3;
  const float* xr = x + (size_t)n * EMBED;
  float s = bi[k];
  for (int j = 0; j < EMBED; j++) s += xr[j] * Wi[j * KLM + k];
  I[idx] = s;
}

// ---------------------------------------------------------------------------
// Landmark pooling: cs = softmax(I, axis=N); cv[b,k,r] = sum_n cs[b,n,k]*Km[b,n,r]
// One block per (b,k) column: 384 threads (one per r)
// ---------------------------------------------------------------------------
__global__ void k_landmark_cv(const float* __restrict__ I, const bf16* __restrict__ Km,
                              float* __restrict__ cv) {
  __shared__ float red[512];
  int b = blockIdx.x >> 2, k = blockIdx.x & 3;
  int tid = threadIdx.x;                          // 384 threads
  const float* Icol = I + (size_t)b * SEQ * KLM + k;

  float m = -3.4e38f;
  for (int n = tid; n < SEQ; n += 384) m = fmaxf(m, Icol[(size_t)n * KLM]);
  red[tid] = m; if (tid < 128) red[384 + tid] = -3.4e38f;
  __syncthreads();
  for (int s = 256; s > 0; s >>= 1) {
    if (tid < s) red[tid] = fmaxf(red[tid], red[tid + s]);
    __syncthreads();
  }
  float cmax = red[0];
  __syncthreads();

  float sum = 0.f;
  for (int n = tid; n < SEQ; n += 384) sum += __expf(Icol[(size_t)n * KLM] - cmax);
  red[tid] = sum; if (tid < 128) red[384 + tid] = 0.f;
  __syncthreads();
  for (int s = 256; s > 0; s >>= 1) {
    if (tid < s) red[tid] += red[tid + s];
    __syncthreads();
  }
  float inv = 1.0f / red[0];

  // thread owns reduced-dim index r = tid
  float acc = 0.f;
  const bf16* Kb = Km + (size_t)b * SEQ * RDIM;
  for (int n = 0; n < SEQ; n++) {
    float w = __expf(Icol[(size_t)n * KLM] - cmax);     // broadcast read
    acc += w * (float)Kb[(size_t)n * RDIM + tid];       // coalesced read
  }
  cv[((size_t)b * KLM + k) * RDIM + tid] = acc * inv;
}

// ---------------------------------------------------------------------------
// ms = softmax(R @ cv^T, axis=k); mcv = ms @ cv  -> bf16 [MTOK, RDIM]
// One wave per token (wave32), cv cached in LDS, __shfl_xor reductions
// ---------------------------------------------------------------------------
__global__ void k_match_mix(const bf16* __restrict__ R, const float* __restrict__ cv,
                            bf16* __restrict__ mcv) {
  __shared__ float cvl[KLM * RDIM];                // 6 KB
  int tid = threadIdx.x, wave = tid >> 5, lane = tid & 31;
  int tok0 = blockIdx.x * 8;                       // 8 waves = 8 tokens per block
  int b = tok0 >> 12;                              // SEQ = 4096
  for (int i = tid; i < KLM * RDIM; i += 256) cvl[i] = cv[(size_t)b * KLM * RDIM + i];
  __syncthreads();

  int token = tok0 + wave;
  const bf16* Rr = R + (size_t)token * RDIM;
  float s0 = 0, s1 = 0, s2 = 0, s3 = 0;
  for (int r = lane; r < RDIM; r += 32) {
    float rv = (float)Rr[r];
    s0 += rv * cvl[r];
    s1 += rv * cvl[RDIM + r];
    s2 += rv * cvl[2 * RDIM + r];
    s3 += rv * cvl[3 * RDIM + r];
  }
  for (int off = 16; off; off >>= 1) {
    s0 += __shfl_xor(s0, off); s1 += __shfl_xor(s1, off);
    s2 += __shfl_xor(s2, off); s3 += __shfl_xor(s3, off);
  }
  float mx = fmaxf(fmaxf(s0, s1), fmaxf(s2, s3));
  float e0 = __expf(s0 - mx), e1 = __expf(s1 - mx), e2 = __expf(s2 - mx), e3 = __expf(s3 - mx);
  float iz = 1.0f / (e0 + e1 + e2 + e3);
  e0 *= iz; e1 *= iz; e2 *= iz; e3 *= iz;

  bf16* out = mcv + (size_t)token * RDIM;
  for (int r = lane; r < RDIM; r += 32) {
    float v = e0 * cvl[r] + e1 * cvl[RDIM + r] + e2 * cvl[2 * RDIM + r] + e3 * cvl[3 * RDIM + r];
    out[r] = (bf16)v;
  }
}

// ---------------------------------------------------------------------------
// Generic bf16 WMMA GEMM: C[M,N] = A[M,K] @ Bt[N,K]^T + bias (+ optional gate)
// 128x128 block tile, BK=32, 8 waves (2M x 4N), wave tile 64x32 (4x2 frags)
// Double-buffered LDS with GLOBAL_LOAD_ASYNC_TO_LDS_B128 when available.
// ---------------------------------------------------------------------------
#define BM 128
#define BN 128
#define BK 32
#define LDT (BK + 8)   // padded LDS row stride: 40 bf16 = 80 B (16B multiple)

typedef union { bf16x8 h[2]; v16bf v; } frag_t;

#if USE_ASYNC
// Each thread issues 2 A-chunks + 2 B-chunks of 16B => 4 async ops / wave / tile
__device__ __forceinline__ void stage_async(const bf16* gA, const bf16* gB,
                                            bf16 (*Asb)[LDT], bf16 (*Bsb)[LDT],
                                            int tid, int Kdim) {
#pragma unroll
  for (int p = 0; p < 2; p++) {
    int lin = p * 256 + tid;
    int r = lin >> 2, c = (lin & 3) << 3;
    async_cp16(gA + (size_t)r * Kdim + c, &Asb[r][c]);
    async_cp16(gB + (size_t)r * Kdim + c, &Bsb[r][c]);
  }
}
#endif

__global__ __launch_bounds__(256)
void k_wmma_gemm(const bf16* __restrict__ A, const bf16* __restrict__ Bt,
                 const float* __restrict__ bias, const bf16* __restrict__ gate,
                 float* __restrict__ Cf, bf16* __restrict__ Cb,
                 int Mdim, int Ndim, int Kdim) {
  __shared__ __attribute__((aligned(16))) bf16 As[2][BM][LDT];
  __shared__ __attribute__((aligned(16))) bf16 Bs[2][BN][LDT];
  int tid  = threadIdx.x;
  int wave = tid >> 5, lane = tid & 31;
  int wm   = wave >> 2, wn = wave & 3;       // 2 x 4 wave grid
  int half = lane >> 4, l16 = lane & 15;
  int bm   = blockIdx.y * BM, bn = blockIdx.x * BN;

  v8f acc[4][2];
#pragma unroll
  for (int i = 0; i < 4; i++)
#pragma unroll
    for (int j = 0; j < 2; j++)
#pragma unroll
      for (int e = 0; e < 8; e++) acc[i][j][e] = 0.0f;

  const bf16* gA = A  + (size_t)bm * Kdim;
  const bf16* gB = Bt + (size_t)bn * Kdim;
  int nk = Kdim / BK;   // K is always a multiple of 32 here

#if USE_ASYNC
  // ---------------- async double-buffered pipeline ----------------
  stage_async(gA, gB, As[0], Bs[0], tid, Kdim);            // tile 0 -> buf 0
  for (int t = 0; t < nk; t++) {
    int cur = t & 1;
    if (t + 1 < nk) {
      stage_async(gA + (size_t)(t + 1) * BK, gB + (size_t)(t + 1) * BK,
                  As[cur ^ 1], Bs[cur ^ 1], tid, Kdim);    // tile t+1 -> other buf
      wait_async_le4();   // async loads retire in order: current tile has landed
    } else {
      wait_async_0();
    }
    __syncthreads();

    // hoist all fragment loads, then issue the 8 WMMAs back-to-back
    frag_t bfr[2], afr[4];
#pragma unroll
    for (int nf = 0; nf < 2; nf++) {
      int n = wn * 32 + nf * 16 + l16;
      bfr[nf].h[0] = *(const bf16x8*)&Bs[cur][n][half * 16];
      bfr[nf].h[1] = *(const bf16x8*)&Bs[cur][n][half * 16 + 8];
    }
#pragma unroll
    for (int mf = 0; mf < 4; mf++) {
      int m = wm * 64 + mf * 16 + l16;
      afr[mf].h[0] = *(const bf16x8*)&As[cur][m][half * 16];
      afr[mf].h[1] = *(const bf16x8*)&As[cur][m][half * 16 + 8];
    }
#pragma unroll
    for (int mf = 0; mf < 4; mf++)
#pragma unroll
      for (int nf = 0; nf < 2; nf++)
        acc[mf][nf] = __builtin_amdgcn_wmma_f32_16x16x32_bf16(
            false, afr[mf].v, false, bfr[nf].v, (short)0, acc[mf][nf], false, false);
    __syncthreads();   // all waves done reading before next tile overwrites
  }
#else
  // ---------------- fallback: synchronous staging ----------------
  for (int t = 0; t < nk; t++) {
    int k0 = t * BK;
#pragma unroll
    for (int p = 0; p < 2; p++) {
      int lin = p * 256 + tid;
      int r = lin >> 2, c = (lin & 3) << 3;
      *(bf16x8*)&As[0][r][c] = *(const bf16x8*)&gA[(size_t)r * Kdim + k0 + c];
      *(bf16x8*)&Bs[0][r][c] = *(const bf16x8*)&gB[(size_t)r * Kdim + k0 + c];
    }
    if (k0 + BK < Kdim) {   // hint next K-tile into cache (global_prefetch_b8)
      __builtin_prefetch(&gA[(size_t)(tid >> 1) * Kdim + k0 + BK], 0, 1);
      __builtin_prefetch(&gB[(size_t)(tid >> 1) * Kdim + k0 + BK], 0, 1);
    }
    __syncthreads();

    frag_t bfr[2], afr[4];
#pragma unroll
    for (int nf = 0; nf < 2; nf++) {
      int n = wn * 32 + nf * 16 + l16;
      bfr[nf].h[0] = *(const bf16x8*)&Bs[0][n][half * 16];
      bfr[nf].h[1] = *(const bf16x8*)&Bs[0][n][half * 16 + 8];
    }
#pragma unroll
    for (int mf = 0; mf < 4; mf++) {
      int m = wm * 64 + mf * 16 + l16;
      afr[mf].h[0] = *(const bf16x8*)&As[0][m][half * 16];
      afr[mf].h[1] = *(const bf16x8*)&As[0][m][half * 16 + 8];
    }
#pragma unroll
    for (int mf = 0; mf < 4; mf++)
#pragma unroll
      for (int nf = 0; nf < 2; nf++)
        acc[mf][nf] = __builtin_amdgcn_wmma_f32_16x16x32_bf16(
            false, afr[mf].v, false, bfr[nf].v, (short)0, acc[mf][nf], false, false);
    __syncthreads();
  }
#endif

  // Epilogue: D layout — VGPR e holds row (e + 8*half), col = l16
#pragma unroll
  for (int mf = 0; mf < 4; mf++) {
#pragma unroll
    for (int nf = 0; nf < 2; nf++) {
      int col = bn + wn * 32 + nf * 16 + l16;
      float bv = bias ? bias[col] : 0.0f;
#pragma unroll
      for (int e = 0; e < 8; e++) {
        int row = bm + wm * 64 + mf * 16 + half * 8 + e;
        size_t idx = (size_t)row * Ndim + col;
        float v = acc[mf][nf][e] + bv;
        if (gate) v *= (float)gate[idx];
        if (Cf) Cf[idx] = v;
        else    Cb[idx] = (bf16)v;
      }
    }
  }
}

// ---------------------------------------------------------------------------
// Host-side orchestration
// ---------------------------------------------------------------------------
extern "C" void kernel_launch(void* const* d_in, const int* in_sizes, int n_in,
                              void* d_out, int out_size, void* d_ws, size_t ws_size,
                              hipStream_t stream) {
  (void)in_sizes; (void)n_in; (void)out_size; (void)ws_size;
  const float* x   = (const float*)d_in[0];
  const float* Wi  = (const float*)d_in[1];
  const float* bi  = (const float*)d_in[2];
  const float* Wk  = (const float*)d_in[3];
  const float* bk  = (const float*)d_in[4];
  const float* Wr  = (const float*)d_in[5];
  const float* br  = (const float*)d_in[6];
  const float* Wv  = (const float*)d_in[7];
  const float* bv  = (const float*)d_in[8];
  const float* Wm  = (const float*)d_in[9];
  const float* bm_ = (const float*)d_in[10];
  const float* Wp  = (const float*)d_in[11];
  const float* bp  = (const float*)d_in[12];
  float* out = (float*)d_out;

  char* ws = (char*)d_ws;
  size_t off = 0;
  auto take = [&](size_t bytes) -> char* {
    char* p = ws + off;
    off += (bytes + 255) & ~(size_t)255;
    return p;
  };

  bf16*  xb   = (bf16*) take((size_t)MTOK * EMBED * 2);   // x in bf16
  bf16*  WkT  = (bf16*) take((size_t)RDIM  * EMBED * 2);
  bf16*  WrT  = (bf16*) take((size_t)RDIM  * EMBED * 2);
  bf16*  WvT  = (bf16*) take((size_t)EMBED * EMBED * 2);
  bf16*  WmT  = (bf16*) take((size_t)EMBED * RDIM  * 2);  // Wm: [RDIM,EMBED] -> [EMBED][RDIM]
  bf16*  WpT  = (bf16*) take((size_t)EMBED * EMBED * 2);
  float* Ibuf = (float*)take((size_t)MTOK * KLM * 4);
  bf16*  Kmb  = (bf16*) take((size_t)MTOK * RDIM * 2);
  bf16*  Rb   = (bf16*) take((size_t)MTOK * RDIM * 2);
  bf16*  Vb   = (bf16*) take((size_t)MTOK * EMBED * 2);
  float* cvb  = (float*)take((size_t)BATCH * KLM * RDIM * 4);
  bf16*  mcvb = (bf16*) take((size_t)MTOK * RDIM * 2);
  bf16*  Hb   = (bf16*) take((size_t)MTOK * EMBED * 2);

  // 1) convert activations + weights to bf16 (weights transposed to [out][in])
  {
    int n = MTOK * EMBED;
    k_f32_to_bf16<<<(n + 255) / 256, 256, 0, stream>>>(x, xb, n);
  }
  k_wt_cvt<<<(EMBED * RDIM  + 255) / 256, 256, 0, stream>>>(Wk, WkT, EMBED, RDIM);
  k_wt_cvt<<<(EMBED * RDIM  + 255) / 256, 256, 0, stream>>>(Wr, WrT, EMBED, RDIM);
  k_wt_cvt<<<(EMBED * EMBED + 255) / 256, 256, 0, stream>>>(Wv, WvT, EMBED, EMBED);
  k_wt_cvt<<<(RDIM  * EMBED + 255) / 256, 256, 0, stream>>>(Wm, WmT, RDIM,  EMBED);
  k_wt_cvt<<<(EMBED * EMBED + 255) / 256, 256, 0, stream>>>(Wp, WpT, EMBED, EMBED);

  // 2) tiny projection I = x@Wi + bi (scalar path, N=4)
  k_proj_I<<<(MTOK * KLM + 255) / 256, 256, 0, stream>>>(x, Wi, bi, Ibuf);

  // 3) big projections via WMMA: Km, R, V (bf16 outputs)
  dim3 blk(256);
  dim3 gR(RDIM  / BN, MTOK / BM);   // (3, 256)
  dim3 gC(EMBED / BN, MTOK / BM);   // (6, 256)
  k_wmma_gemm<<<gR, blk, 0, stream>>>(xb, WkT, bk, nullptr, nullptr, Kmb,
                                      MTOK, RDIM, EMBED);
  k_wmma_gemm<<<gR, blk, 0, stream>>>(xb, WrT, br, nullptr, nullptr, Rb,
                                      MTOK, RDIM, EMBED);
  k_wmma_gemm<<<gC, blk, 0, stream>>>(xb, WvT, bv, nullptr, nullptr, Vb,
                                      MTOK, EMBED, EMBED);

  // 4) landmark context vectors (softmax over tokens + pooling)
  k_landmark_cv<<<BATCH * KLM, 384, 0, stream>>>(Ibuf, Kmb, cvb);

  // 5) match tokens to landmarks, mix -> mcv (bf16)
  k_match_mix<<<MTOK / 8, 256, 0, stream>>>(Rb, cvb, mcvb);

  // 6) H = (mcv@Wm + bm) * V  (gate fused in epilogue, bf16 out)
  k_wmma_gemm<<<gC, blk, 0, stream>>>(mcvb, WmT, bm_, Vb, nullptr, Hb,
                                      MTOK, EMBED, RDIM);

  // 7) out = H@Wp + bp  (f32 straight to d_out)
  k_wmma_gemm<<<gC, blk, 0, stream>>>(Hb, WpT, bp, nullptr, out, nullptr,
                                      MTOK, EMBED, EMBED);
}
